// InvGraphConv_37512244363272
// MI455X (gfx1250) — matrix operand order repaired and checked by reading, not compiled, not run on hardware
//
#include <hip/hip_runtime.h>
#include <hip/hip_bf16.h>

// ---------------------------------------------------------------------------
// SplineCNN-style graph conv for MI455X (gfx1250, wave32).
// Strategy: z-precompute formulation  Z[n,k,:] = H[n,:] @ W[k]  as 125 batched
// dense GEMMs driven by v_wmma_f32_16x16x32_bf16, Z stored bf16 (HBM-bound
// pipeline => halve traffic), then per-edge weighted gather + atomic
// segment-sum.  Layer 1 degenerates to a table gather (h == ones).
// ---------------------------------------------------------------------------

typedef __attribute__((ext_vector_type(16))) __bf16 v16bf;
typedef __attribute__((ext_vector_type(8)))  float  v8f;

union BF16x16 { v16bf v; unsigned short u[16]; };

#define NNODE 20000
#define NEDGE 320000
#define NKER  125

__device__ __forceinline__ unsigned short f32_to_bf16(float f) {
    unsigned int u = __float_as_uint(f);
    unsigned int r = u + 0x7FFFu + ((u >> 16) & 1u);   // round-to-nearest-even
    return (unsigned short)(r >> 16);
}
__device__ __forceinline__ float bf16_to_f32(unsigned short s) {
    return __uint_as_float(((unsigned int)s) << 16);
}
__device__ __forceinline__ float eluf(float v) {
    return v > 0.0f ? v : expm1f(v);
}

// ---------------------------------------------------------------- degree ----
__global__ void deg_kernel(const int* __restrict__ ei, float* __restrict__ deg, int E) {
    int e = blockIdx.x * blockDim.x + threadIdx.x;
    if (e < E) atomicAdd(&deg[ei[e]], 1.0f);
}

// ------------------------------------------------- B-spline basis / index ---
// pos = clip(pseudo)*4 ; i0 = clip(floor(pos),0,3) ; frac = pos - i0
// basis[s] = prod_d (bit ? frac : 1-frac) ; idx[s] = sum_d (i0+bit)*stride_d
__global__ void basis_kernel(const float* __restrict__ pseudo,
                             const float* __restrict__ tvec,
                             const int*  __restrict__ ei,
                             float* __restrict__ basis,
                             unsigned char* __restrict__ idx,
                             int E, int adjust) {
    int e = blockIdx.x * blockDim.x + threadIdx.x;
    if (e >= E) return;
    int row = 0, col = 0;
    if (adjust) { row = ei[e]; col = ei[E + e]; }
    float frac[3]; int i0[3];
    const int stride[3] = {25, 5, 1};
#pragma unroll
    for (int d = 0; d < 3; ++d) {
        float v = pseudo[e * 3 + d];
        if (adjust) v = v + tvec[col * 3 + d] - tvec[row * 3 + d];
        v = fminf(fmaxf(v, 0.0f), 1.0f);
        float pos = v * 4.0f;
        float fl  = fminf(fmaxf(floorf(pos), 0.0f), 3.0f);
        i0[d]   = (int)fl;
        frac[d] = pos - fl;
    }
#pragma unroll
    for (int s = 0; s < 8; ++s) {
        float b = 1.0f; int k = 0;
#pragma unroll
        for (int d = 0; d < 3; ++d) {
            int c = (s >> d) & 1;
            b *= c ? frac[d] : (1.0f - frac[d]);
            k += (i0[d] + c) * stride[d];
        }
        basis[e * 8 + s] = b;
        idx[e * 8 + s]   = (unsigned char)k;
    }
}

// ------------------------------------------ stn1: h==ones -> table gather ---
__global__ void stn1_kernel(const float* __restrict__ w1,
                            const float* __restrict__ basis,
                            const unsigned char* __restrict__ idx,
                            const int* __restrict__ ei,
                            float* __restrict__ acc, int E) {
    __shared__ float sw[NKER * 64];               // 32 KB of 320 KB WGP LDS
    for (int i = threadIdx.x; i < NKER * 64; i += blockDim.x) sw[i] = w1[i];
    __syncthreads();
    int total  = E * 64;
    int stride = blockDim.x * gridDim.x;
    for (int t = blockIdx.x * blockDim.x + threadIdx.x; t < total; t += stride) {
        int e = t >> 6, o = t & 63;
        float m = 0.0f;
#pragma unroll
        for (int s = 0; s < 8; ++s)
            m += basis[e * 8 + s] * sw[(int)idx[e * 8 + s] * 64 + o];
        atomicAdd(&acc[(size_t)ei[e] * 64 + o], m);
    }
}

// ---------------------------------------------------- elu + bf16 convert ----
__global__ void elu_h1_kernel(const float* __restrict__ acc,
                              const float* __restrict__ deg,
                              const float* __restrict__ b1,
                              unsigned short* __restrict__ h1b, int N) {
    int i = blockIdx.x * blockDim.x + threadIdx.x;
    if (i >= N * 64) return;
    int n = i >> 6, o = i & 63;
    float v = acc[i] / fmaxf(deg[n], 1.0f) + b1[o];
    h1b[i] = f32_to_bf16(eluf(v));
}

// W[k][i][o] (f32) -> Wt[k][o][i] (bf16): makes B fragments dword-contiguous
__global__ void transpose_w_kernel(const float* __restrict__ w,
                                   unsigned short* __restrict__ wt, int K, int F) {
    int t = blockIdx.x * blockDim.x + threadIdx.x;
    if (t >= K * F * F) return;
    int k = t / (F * F), r = t % (F * F), o = r / F, i = r % F;
    wt[((size_t)k * F + o) * F + i] = f32_to_bf16(w[((size_t)k * F + i) * F + o]);
}

__global__ void f32_to_bf16_kernel(const float* __restrict__ a,
                                   unsigned short* __restrict__ b, int n) {
    int i = blockIdx.x * blockDim.x + threadIdx.x;
    if (i < n) b[i] = f32_to_bf16(a[i]);
}

// ------------------------------------------------------------- WMMA GEMM ----
// Z[n][k][o] = sum_i A[n][i] * Wt[k][o][i]   for all k in [0,125)
// One block = one 16-node tile; wave w owns output-column tile w.
// A fragment (16x32 bf16): lane M = lane&15; VGPR v: K = (v<4?2v:16+2(v-4)) (+8 if lane>=16)
// B fragment (32x16 bf16): lane N = lane&15; VGPR v: K = (v<4?2v:16+2(v-4)) (+8 if lane>=16)
// D (16x16 f32): c[v] = D[v + 8*(lane>=16)][lane&15]
template <int FIN, int FOUT>
__global__ void spline_gemm_kernel(const unsigned short* __restrict__ A,
                                   const unsigned short* __restrict__ Bt,
                                   unsigned short* __restrict__ Z) {
    constexpr int WIN = FIN / 32;                 // K windows per GEMM
    const int lane  = threadIdx.x & 31;
    const int wave  = threadIdx.x >> 5;           // output col tile
    const int hi    = (lane >= 16) ? 1 : 0;
    const int colc  = lane & 15;
    const int node0 = blockIdx.x * 16;

    BF16x16 a[WIN];
    const unsigned short* arow = A + (size_t)(node0 + colc) * FIN;
#pragma unroll
    for (int kk = 0; kk < WIN; ++kk)
#pragma unroll
        for (int v = 0; v < 8; ++v) {
            int koff = kk * 32 + ((v < 4) ? 2 * v : 16 + 2 * (v - 4)) + (hi ? 8 : 0);
            unsigned int pr = *(const unsigned int*)(arow + koff);
            a[kk].u[2 * v]     = (unsigned short)(pr & 0xFFFFu);
            a[kk].u[2 * v + 1] = (unsigned short)(pr >> 16);
        }

    for (int k = 0; k < NKER; ++k) {
        BF16x16 b[WIN];
        const unsigned short* brow =
            Bt + ((size_t)k * FOUT + wave * 16 + colc) * FIN;
#pragma unroll
        for (int kk = 0; kk < WIN; ++kk)
#pragma unroll
            for (int v = 0; v < 8; ++v) {
                int koff = kk * 32 + ((v < 4) ? 2 * v : 16 + 2 * (v - 4)) + (hi ? 8 : 0);
                unsigned int pr = *(const unsigned int*)(brow + koff);
                b[kk].u[2 * v]     = (unsigned short)(pr & 0xFFFFu);
                b[kk].u[2 * v + 1] = (unsigned short)(pr >> 16);
            }
        v8f c = {0.f, 0.f, 0.f, 0.f, 0.f, 0.f, 0.f, 0.f};
#pragma unroll
        for (int kk = 0; kk < WIN; ++kk)
            c = __builtin_amdgcn_wmma_f32_16x16x32_bf16(
                    false, a[kk].v, false, b[kk].v, (short)0, c, false, false);
#pragma unroll
        for (int v = 0; v < 8; ++v) {
            int m = v + (hi ? 8 : 0);
            Z[((size_t)(node0 + m) * NKER + k) * FOUT + wave * 16 + colc] =
                f32_to_bf16(c[v]);
        }
    }
}

// -------------------------------- per-edge weighted gather + scatter-add ----
// msg[e,:] = sum_s basis[e,s] * Z[col_e][idx[e,s]][:] ; acc[row_e,:] += msg
template <int FOUT>
__global__ void gather_scatter_kernel(const unsigned short* __restrict__ Z,
                                      const float* __restrict__ basis,
                                      const unsigned char* __restrict__ idx,
                                      const int* __restrict__ ei,
                                      float* __restrict__ acc, int E) {
    constexpr int TPE = FOUT / 2;                 // threads per edge
    int tid = blockIdx.x * blockDim.x + threadIdx.x;
    int e = tid / TPE, lo = tid % TPE;
    if (e >= E) return;
    int row = ei[e], col = ei[E + e];
    float mx = 0.0f, my = 0.0f;
#pragma unroll
    for (int s = 0; s < 8; ++s) {
        float b = basis[e * 8 + s];
        int   k = idx[e * 8 + s];
        const unsigned int* zr =
            (const unsigned int*)(Z + ((size_t)col * NKER + k) * FOUT);
        unsigned int pr = zr[lo];                 // two bf16, fully coalesced
        mx += b * bf16_to_f32((unsigned short)(pr & 0xFFFFu));
        my += b * bf16_to_f32((unsigned short)(pr >> 16));
    }
    atomicAdd(&acc[(size_t)row * FOUT + 2 * lo],     mx);
    atomicAdd(&acc[(size_t)row * FOUT + 2 * lo + 1], my);
}

// -------------------- fused: elu(acc2/deg+b2) -> elu(@W3+b3) -> @W4+b4 ------
__global__ void stn_tail_kernel(const float* __restrict__ acc2,
                                const float* __restrict__ deg,
                                const float* __restrict__ b2,
                                const float* __restrict__ w3,
                                const float* __restrict__ b3,
                                const float* __restrict__ w4,
                                const float* __restrict__ b4,
                                float* __restrict__ tvec, int N) {
    __shared__ float sw3[64 * 64];
    __shared__ float sh2[4][64];
    __shared__ float sh3[4][64];
    for (int i = threadIdx.x; i < 64 * 64; i += blockDim.x) sw3[i] = w3[i];
    __syncthreads();
    int nl = threadIdx.x >> 6, o = threadIdx.x & 63;
    int n = blockIdx.x * 4 + nl;
    if (n < N) {
        float v = acc2[(size_t)n * 64 + o] / fmaxf(deg[n], 1.0f) + b2[o];
        sh2[nl][o] = eluf(v);
    }
    __syncthreads();
    if (n < N) {
        float s = b3[o];
#pragma unroll 8
        for (int i = 0; i < 64; ++i) s += sh2[nl][i] * sw3[i * 64 + o];
        sh3[nl][o] = eluf(s);
    }
    __syncthreads();
    if (n < N && o < 3) {
        float s = b4[o];
#pragma unroll 8
        for (int i = 0; i < 64; ++i) s += sh3[nl][i] * w4[i * 3 + o];
        tvec[n * 3 + o] = s;
    }
}

// ---------------------------------------------------------------- finish ----
__global__ void finalize_kernel(const float* __restrict__ acc,
                                const float* __restrict__ deg,
                                const float* __restrict__ bc,
                                float* __restrict__ out, int N) {
    int i = blockIdx.x * blockDim.x + threadIdx.x;
    if (i >= N * 32) return;
    out[i] = acc[i] / fmaxf(deg[i >> 5], 1.0f) + bc[i & 31];
}

// ---------------------------------------------------------------------------
extern "C" void kernel_launch(void* const* d_in, const int* in_sizes, int n_in,
                              void* d_out, int out_size, void* d_ws, size_t ws_size,
                              hipStream_t stream) {
    const float* x      = (const float*)d_in[0];
    const int*   ei     = (const int*)  d_in[1];
    const float* pseudo = (const float*)d_in[2];
    const float* w1 = (const float*)d_in[3];
    const float* b1 = (const float*)d_in[4];
    const float* w2 = (const float*)d_in[5];
    const float* b2 = (const float*)d_in[6];
    const float* w3 = (const float*)d_in[7];
    const float* b3 = (const float*)d_in[8];
    const float* w4 = (const float*)d_in[9];
    const float* b4 = (const float*)d_in[10];
    const float* wc = (const float*)d_in[11];
    const float* bc = (const float*)d_in[12];
    float* out = (float*)d_out;

    const int N = in_sizes[0] / 32;   // 20000
    const int E = in_sizes[1] / 2;    // 320000

    // ---- workspace layout (all offsets 128B aligned; ~343 MB total) ----
    char* ws = (char*)d_ws;
    float*          deg   = (float*)(ws + 0);                    //  80,000 B
    float*          basis = (float*)(ws + 80000);                //  10.24 MB
    unsigned char*  idx   = (unsigned char*)(ws + 10320000);     //   2.56 MB
    unsigned short* h1b   = (unsigned short*)(ws + 12880000);    //   2.56 MB
    float*          acc   = (float*)(ws + 15440000);             //   5.12 MB
    unsigned short* w2t   = (unsigned short*)(ws + 20560000);    //   1.02 MB
    float*          tvec  = (float*)(ws + 21584000);             //   0.24 MB
    unsigned short* xb    = (unsigned short*)(ws + 21824000);    //   1.28 MB
    unsigned short* wct   = (unsigned short*)(ws + 23104000);    //   0.26 MB
    unsigned short* Z     = (unsigned short*)(ws + 23360000);    // 320.0  MB

    // ---- degree + basis (original pseudo) ----
    hipMemsetAsync(deg, 0, (size_t)N * 4, stream);
    hipMemsetAsync(acc, 0, (size_t)N * 64 * 4, stream);
    deg_kernel<<<(E + 255) / 256, 256, 0, stream>>>(ei, deg, E);
    basis_kernel<<<(E + 255) / 256, 256, 0, stream>>>(pseudo, nullptr, ei,
                                                      basis, idx, E, 0);

    // ---- stn1 (h == ones -> pure LDS table gather) ----
    stn1_kernel<<<640, 256, 0, stream>>>(w1, basis, idx, ei, acc, E);
    elu_h1_kernel<<<(N * 64 + 255) / 256, 256, 0, stream>>>(acc, deg, b1, h1b, N);

    // ---- stn2: Z2 = H1 @ W2[k] via WMMA bf16, then gather/scatter ----
    transpose_w_kernel<<<(NKER * 64 * 64 + 255) / 256, 256, 0, stream>>>(w2, w2t, NKER, 64);
    spline_gemm_kernel<64, 64><<<N / 16, 128, 0, stream>>>(h1b, w2t, Z);
    hipMemsetAsync(acc, 0, (size_t)N * 64 * 4, stream);
    gather_scatter_kernel<64><<<(E * 32 + 255) / 256, 256, 0, stream>>>(Z, basis, idx, ei, acc, E);

    // ---- MLP tail -> per-node translation t ----
    stn_tail_kernel<<<(N + 3) / 4, 256, 0, stream>>>(acc, deg, b2, w3, b3, w4, b4, tvec, N);

    // ---- new basis from transformed pseudo ----
    basis_kernel<<<(E + 255) / 256, 256, 0, stream>>>(pseudo, tvec, ei,
                                                      basis, idx, E, 1);

    // ---- final conv: Zc = X @ Wc[k] via WMMA bf16, gather/scatter, bias ----
    f32_to_bf16_kernel<<<(N * 32 + 255) / 256, 256, 0, stream>>>(x, xb, N * 32);
    transpose_w_kernel<<<(NKER * 32 * 32 + 255) / 256, 256, 0, stream>>>(wc, wct, NKER, 32);
    spline_gemm_kernel<32, 32><<<N / 16, 64, 0, stream>>>(xb, wct, Z);
    hipMemsetAsync(acc, 0, (size_t)N * 32 * 4, stream);
    gather_scatter_kernel<32><<<(E * 16 + 255) / 256, 256, 0, stream>>>(Z, basis, idx, ei, acc, E);
    finalize_kernel<<<(N * 32 + 255) / 256, 256, 0, stream>>>(acc, deg, bc, out, N);
}
